// MultiHeadAttention_27101243638241
// MI455X (gfx1250) — matrix operand order repaired and checked by reading, not compiled, not run on hardware
//
#include <hip/hip_runtime.h>
#include <stdint.h>

// ---------------------------------------------------------------------------
// MI455X (gfx1250) multi-head attention, all matmuls via v_wmma_f32_16x16x32_bf16
// B=4, S=2048, D=1024, H=16, DK=64
// ---------------------------------------------------------------------------

#define B_  4
#define S_  2048
#define D_  1024
#define H_  16
#define DK_ 64

typedef __attribute__((ext_vector_type(16))) __bf16 v16bf;
typedef __attribute__((ext_vector_type(8)))  float  v8f;
typedef __attribute__((ext_vector_type(4)))  unsigned int u32x4;
typedef __attribute__((ext_vector_type(8)))  int          i32x8;
typedef __attribute__((ext_vector_type(4)))  int          i32x4;

union AFrag {
    v16bf    v;
    float4   f4[2];
    uint16_t u16[16];
};

__device__ inline uint16_t f2bf(float f) {
    union { float f; uint32_t u; } x; x.f = f;
    uint32_t r = x.u + 0x7FFFu + ((x.u >> 16) & 1u);   // round-to-nearest-even
    return (uint16_t)(r >> 16);
}

__device__ inline v8f wmma_bf16(const v16bf& a, const v16bf& b, v8f c) {
    // 8 args: (neg_a, A, neg_b, B, c_mod, C, reuse_a, reuse_b)
    return __builtin_amdgcn_wmma_f32_16x16x32_bf16(false, a, false, b,
                                                   (short)0, c, false, false);
}

#if __has_builtin(__builtin_amdgcn_tensor_load_to_lds) && \
    __has_builtin(__builtin_amdgcn_s_wait_tensorcnt)
#define USE_TDM 1
// 2D bf16 tile DMA: global (strideElems elems/row) -> LDS, via Tensor Data Mover.
// D# per CDNA5 ISA ch.8: group0 {count=1, lds_addr, global_addr57, type=2},
// group1 {data_size=2B, tensor_dim0/1, tile_dim0/1, dim0_stride}.
// This toolchain exposes the 6-arg builtin (clang-23 / therock-10.0 form).
__device__ inline void tdm_load_2d_bf16(uint32_t ldsAddr, const void* gaddr,
                                        uint32_t tileW, uint32_t tileH,
                                        uint32_t tensorW, uint32_t tensorH,
                                        uint32_t strideElems) {
    uint64_t ga = (uint64_t)(uintptr_t)gaddr;
    u32x4 g0;
    g0[0] = 1u;                                            // count=1 (user mode)
    g0[1] = ldsAddr;                                       // [63:32] lds_addr
    g0[2] = (uint32_t)ga;                                  // global_addr[31:0]
    g0[3] = (uint32_t)((ga >> 32) & 0x1FFFFFFu)            // global_addr[56:32]
          | (2u << 30);                                    // type=2 ("image")
    i32x8 g1;
    g1[0] = (int)(1u << 16);                               // data_size=1 -> 2 bytes
    g1[1] = (int)((tensorW & 0xFFFFu) << 16);              // tensor_dim0[15:0]
    g1[2] = (int)((tensorW >> 16) | ((tensorH & 0xFFFFu) << 16));
    g1[3] = (int)((tensorH >> 16) | (tileW << 16));        // tile_dim0
    g1[4] = (int)tileH;                                    // tile_dim1 (tile_dim2=0)
    g1[5] = (int)strideElems;                              // tensor_dim0_stride lo
    g1[6] = 0;
    g1[7] = 0;
    i32x4 z4 = {0, 0, 0, 0};
    i32x8 z8 = {0, 0, 0, 0, 0, 0, 0, 0};
    __builtin_amdgcn_tensor_load_to_lds(g0, g1, z4, z4, z8, 0);
}
#else
#define USE_TDM 0
#endif

// ---------------------------------------------------------------------------
// GEMM: Y[m,n] = sum_k X[m,k] * W[n,k]   (y = x @ W^T)
// Block tile 128(M) x 128(N), K-step 32; 8 waves = 4(M) x 2(N), each wave a
// 32x64 tile via 2x4 WMMA accumulators (8 back-to-back WMMAs per K-step).
// XT: float (convert->bf16) or uint16_t (bf16 bits). YT: uint16_t(bf16)/float.
// ---------------------------------------------------------------------------
template <typename XT, typename YT>
__global__ __launch_bounds__(256) void gemm_xwt_kernel(
    const XT* __restrict__ X, const float* __restrict__ W, YT* __restrict__ Y,
    int M, int N, int K) {
    __shared__ __align__(16) uint16_t As[128 * 32];
    __shared__ __align__(16) uint16_t Bs[128 * 32];

    const int tid    = threadIdx.x;
    const int m0     = blockIdx.x * 128;
    const int n0     = blockIdx.y * 128;
    const int lane   = tid & 31;
    const int wid    = tid >> 5;
    const int lh     = lane & 15;
    const int hiHalf = lane >> 4;
    const int waveM  = wid & 3;   // 4 waves along M (32 rows each)
    const int waveN  = wid >> 2;  // 2 waves along N (64 cols each)

    v8f acc[2][4] = {};

    for (int k0 = 0; k0 < K; k0 += 32) {
        // ---- cooperative fill of As: 128x32 bf16 (4096 elems, 16/thread) ----
#pragma unroll
        for (int it = 0; it < 4; ++it) {
            int i = tid * 4 + it * 1024;
            int r = i >> 5, c = i & 31;
            if constexpr (sizeof(XT) == 4) {
                float4 x4 = *(const float4*)(X + (size_t)(m0 + r) * K + k0 + c);
                As[i + 0] = f2bf(x4.x); As[i + 1] = f2bf(x4.y);
                As[i + 2] = f2bf(x4.z); As[i + 3] = f2bf(x4.w);
            } else {
                *(uint2*)&As[i] =
                    *(const uint2*)((const uint16_t*)X + (size_t)(m0 + r) * K + k0 + c);
            }
        }
        // ---- cooperative fill of Bs: 128x32 bf16 (4096 elems, 16/thread) ----
#pragma unroll
        for (int it = 0; it < 4; ++it) {
            int i = tid * 4 + it * 1024;
            int r = i >> 5, c = i & 31;
            float4 w4 = *(const float4*)(W + (size_t)(n0 + r) * K + k0 + c);
            Bs[i + 0] = f2bf(w4.x); Bs[i + 1] = f2bf(w4.y);
            Bs[i + 2] = f2bf(w4.z); Bs[i + 3] = f2bf(w4.w);
        }
        // ---- prefetch next K-slab (global_prefetch_b8) ----
        if (k0 + 32 < K) {
            int i = tid * 4;
            int r = i >> 5, c = i & 31;
            __builtin_prefetch(X + (size_t)(m0 + r) * K + (k0 + 32) + c, 0, 1);
            __builtin_prefetch(W + (size_t)(n0 + r) * K + (k0 + 32) + c, 0, 1);
        }
        __syncthreads();

        // ---- fragments from LDS (ISA §7.12.2 layouts) ----
        AFrag a[2], b[4];
#pragma unroll
        for (int mi = 0; mi < 2; ++mi) {
            int m  = waveM * 32 + mi * 16 + lh;
            int kA = hiHalf * 8;
            a[mi].f4[0] = *(const float4*)&As[m * 32 + kA];        // K kA..kA+7
            a[mi].f4[1] = *(const float4*)&As[m * 32 + kA + 16];   // K kA+16..kA+23
        }
#pragma unroll
        for (int ni = 0; ni < 4; ++ni) {
            int n  = waveN * 64 + ni * 16 + lh;
            int kB = hiHalf * 8;
            b[ni].f4[0] = *(const float4*)&Bs[n * 32 + kB];        // K kB..kB+7
            b[ni].f4[1] = *(const float4*)&Bs[n * 32 + kB + 16];   // K kB+16..kB+23
        }
#pragma unroll
        for (int mi = 0; mi < 2; ++mi)
#pragma unroll
            for (int ni = 0; ni < 4; ++ni)
                acc[mi][ni] = wmma_bf16(a[mi].v, b[ni].v, acc[mi][ni]);
        __syncthreads();
    }

    // ---- epilogue: C layout lane l, VGPR r -> M = r + 8*(l>=16), N = l&15 ----
#pragma unroll
    for (int mi = 0; mi < 2; ++mi)
#pragma unroll
        for (int ni = 0; ni < 4; ++ni)
#pragma unroll
            for (int r = 0; r < 8; ++r) {
                int m = m0 + waveM * 32 + mi * 16 + r + hiHalf * 8;
                int n = n0 + waveN * 64 + ni * 16 + lh;
                float val = acc[mi][ni][r];
                if constexpr (sizeof(YT) == 2)
                    ((uint16_t*)Y)[(size_t)m * N + n] = f2bf(val);
                else
                    Y[(size_t)m * N + n] = val;
            }
}

// ---------------------------------------------------------------------------
// Flash attention: one wave per 16-query tile; online softmax; causal mask.
// Q/K/V/Ctx are bf16 [B,S,D]. V tiles staged to LDS by the Tensor Data Mover
// (async, overlapped with the QK^T WMMAs), waited with s_wait_tensorcnt.
// ---------------------------------------------------------------------------
__global__ __launch_bounds__(32) void attn_kernel(
    const uint16_t* __restrict__ Qb, const uint16_t* __restrict__ Kb,
    const uint16_t* __restrict__ Vb, uint16_t* __restrict__ Ctx) {
    __shared__ __align__(16) uint16_t Pl[16 * 32];   // P tile (16 q x 32 keys)
    __shared__ __align__(16) uint16_t Vl[32 * 64];   // V tile (32 keys x 64 d)

    const int lane   = threadIdx.x & 31;
    const int lh     = lane & 15;
    const int hiHalf = lane >> 4;

    const int QT   = S_ / 16;
    int tile  = blockIdx.x;
    int qt    = tile % QT;
    int bh    = tile / QT;
    int h     = bh % H_;
    int b     = bh / H_;
    int qBase = qt * 16;

    // ---- Q fragments (A matrix 16x64 -> two 16x32 fragments), kept in regs ----
    AFrag qf[2];
    {
        const uint16_t* Qrow =
            Qb + ((size_t)(b * S_ + qBase + lh)) * D_ + h * DK_;
        int kA = hiHalf * 8;
#pragma unroll
        for (int j = 0; j < 2; ++j) {
            qf[j].f4[0] = *(const float4*)(Qrow + j * 32 + kA);
            qf[j].f4[1] = *(const float4*)(Qrow + j * 32 + kA + 16);
        }
    }

    float mrow[8], lrow[8];
    v8f   o[4] = {};
#pragma unroll
    for (int r = 0; r < 8; ++r) { mrow[r] = -1e30f; lrow[r] = 0.f; }

    for (int kb = 0; kb <= qBase + 15; kb += 32) {
        // ---- stage V tile (32 keys x 64 d) ----
#if USE_TDM
        // Async DMA via TDM; rows past the end of the sequence read as zero
        // (OOB returns zero) and are killed by P==0 anyway.
        tdm_load_2d_bf16((uint32_t)(uintptr_t)Vl,
                         Vb + ((size_t)(b * S_ + kb)) * D_ + h * DK_,
                         /*tileW=*/64, /*tileH=*/32,
                         /*tensorW=*/64, /*tensorH=*/(uint32_t)(S_ - kb),
                         /*strideElems=*/D_);
#else
        {
            int kv = kb + lane; if (kv > S_ - 1) kv = S_ - 1;
            const uint4* src =
                (const uint4*)(Vb + ((size_t)(b * S_ + kv)) * D_ + h * DK_);
            uint4* dst = (uint4*)&Vl[lane * 64];
#pragma unroll
            for (int i = 0; i < 8; ++i) dst[i] = src[i];
        }
#endif

        // ---- scores: two 16x16 tiles (keys kb..kb+15, kb+16..kb+31) ----
        v8f s[2];
#pragma unroll
        for (int t = 0; t < 2; ++t) {
            v8f c = {};
            int kg = kb + t * 16 + lh;               // B frag: N = key
            int kload = kg; if (kload > S_ - 1) kload = S_ - 1;
            const uint16_t* Krow =
                Kb + ((size_t)(b * S_ + kload)) * D_ + h * DK_;
            int kB = hiHalf * 8;
#pragma unroll
            for (int j = 0; j < 2; ++j) {            // d = 0..31, 32..63
                AFrag bf_;
                bf_.f4[0] = *(const float4*)(Krow + j * 32 + kB);
                bf_.f4[1] = *(const float4*)(Krow + j * 32 + kB + 16);
                c = wmma_bf16(qf[j].v, bf_.v, c);
            }
            s[t] = c;
        }

        // ---- scale + causal mask + online softmax ----
        float alpha[8];
#pragma unroll
        for (int r = 0; r < 8; ++r) {
            int qg = qBase + r + hiHalf * 8;
#pragma unroll
            for (int t = 0; t < 2; ++t) {
                int   kg  = kb + t * 16 + lh;
                float val = s[t][r] * 0.125f;        // 1/sqrt(64)
                s[t][r]   = (kg > qg) ? -1e30f : val;
            }
            float v = fmaxf(s[0][r], s[1][r]);
#pragma unroll
            for (int off = 8; off >= 1; off >>= 1)
                v = fmaxf(v, __shfl_xor(v, off, 32));   // half-wave row max
            float mnew = fmaxf(mrow[r], v);
            alpha[r]   = __expf(mrow[r] - mnew);

            float p0 = __expf(s[0][r] - mnew);
            float p1 = __expf(s[1][r] - mnew);
            Pl[(r + 8 * hiHalf) * 32 + lh]      = f2bf(p0);
            Pl[(r + 8 * hiHalf) * 32 + 16 + lh] = f2bf(p1);

            float ps = p0 + p1;
#pragma unroll
            for (int off = 8; off >= 1; off >>= 1)
                ps += __shfl_xor(ps, off, 32);          // half-wave row sum
            lrow[r] = lrow[r] * alpha[r] + ps;
            mrow[r] = mnew;
        }
#pragma unroll
        for (int f = 0; f < 4; ++f)
#pragma unroll
            for (int r = 0; r < 8; ++r) o[f][r] *= alpha[r];

#if USE_TDM
        __builtin_amdgcn_s_wait_tensorcnt(0);  // V tile landed in LDS
#endif
        __syncthreads();   // Pl / Vl visible across lanes

        // ---- P @ V : A frag from Pl, 4 B frags from Vl (d tiles of 16) ----
        AFrag pa;
        {
            int kA = hiHalf * 8;
            pa.f4[0] = *(const float4*)&Pl[lh * 32 + kA];
            pa.f4[1] = *(const float4*)&Pl[lh * 32 + kA + 16];
        }
#pragma unroll
        for (int f = 0; f < 4; ++f) {
            AFrag vf;
#pragma unroll
            for (int e = 0; e < 16; ++e)
                vf.u16[e] = Vl[(hiHalf * 8 + (e & 7) + ((e >> 3) << 4)) * 64 + f * 16 + lh];
            o[f] = wmma_bf16(pa.v, vf.v, o[f]);
        }

        __syncthreads();   // protect Pl/Vl before next iteration overwrites
    }

    // ---- normalize and store context (bf16) ----
#pragma unroll
    for (int f = 0; f < 4; ++f)
#pragma unroll
        for (int r = 0; r < 8; ++r) {
            int   m   = qBase + r + 8 * hiHalf;
            int   d   = f * 16 + lh;
            float val = o[f][r] / lrow[r];
            Ctx[((size_t)(b * S_ + m)) * D_ + h * DK_ + d] = f2bf(val);
        }
}

// ---------------------------------------------------------------------------
extern "C" void kernel_launch(void* const* d_in, const int* in_sizes, int n_in,
                              void* d_out, int out_size, void* d_ws, size_t ws_size,
                              hipStream_t stream) {
    const float* q  = (const float*)d_in[0];
    const float* k  = (const float*)d_in[1];
    const float* v  = (const float*)d_in[2];
    // d_in[3] = causal mask; computed analytically in attn_kernel
    const float* Wq = (const float*)d_in[4];
    const float* Wk = (const float*)d_in[5];
    const float* Wv = (const float*)d_in[6];
    const float* Wo = (const float*)d_in[7];
    float*       out = (float*)d_out;

    const int M = B_ * S_;   // 8192
    const int N = D_;        // 1024
    const int K = D_;        // 1024

    size_t planeElems = (size_t)M * D_;          // 8.4M elems, 16MB as bf16
    uint16_t* Qb = (uint16_t*)d_ws;
    uint16_t* Kb = Qb + planeElems;
    uint16_t* Vb = Kb + planeElems;
    uint16_t* Cx = Vb + planeElems;              // total 64MB of d_ws

    dim3 gg(M / 128, N / 128), gb(256);
    gemm_xwt_kernel<float, uint16_t><<<gg, gb, 0, stream>>>(q, Wq, Qb, M, N, K);
    gemm_xwt_kernel<float, uint16_t><<<gg, gb, 0, stream>>>(k, Wk, Kb, M, N, K);
    gemm_xwt_kernel<float, uint16_t><<<gg, gb, 0, stream>>>(v, Wv, Vb, M, N, K);

    attn_kernel<<<B_ * H_ * (S_ / 16), 32, 0, stream>>>(Qb, Kb, Vb, Cx);

    gemm_xwt_kernel<uint16_t, float><<<gg, gb, 0, stream>>>(Cx, Wo, out, M, N, K);
}
